// MoonViTVisionTower_48902497632990
// MI455X (gfx1250) — matrix operand order repaired and checked by reading, not compiled, not run on hardware
//
#include <hip/hip_runtime.h>
#include <math.h>

// ---------------------------------------------------------------------------
// Types for CDNA5 WMMA (wave32): bf16 16x16x32, f32 accumulate
// ---------------------------------------------------------------------------
typedef __bf16 v16bf __attribute__((ext_vector_type(16)));
typedef float  v8f   __attribute__((ext_vector_type(8)));
typedef float  f32x4 __attribute__((ext_vector_type(4)));
typedef int    v4i   __attribute__((ext_vector_type(4)));

union AF { v16bf v; f32x4 q[2]; };

__device__ __forceinline__ unsigned short f2bf(float f) {
  unsigned int u = __float_as_uint(f);
  unsigned int r = u + 0x7FFFu + ((u >> 16) & 1u);   // round-to-nearest-even
  return (unsigned short)(r >> 16);
}

// ---------------------------------------------------------------------------
// Async global->LDS staging (CDNA5 GLOBAL_LOAD_ASYNC_TO_LDS_B128, ASYNCcnt).
// Prototype (from hipcc diagnostic): (v4i AS1*, v4i AS3*, imm offset, imm cpol)
// Guarded: falls back to a synchronous copy if the builtin is absent.
// ---------------------------------------------------------------------------
#if __has_builtin(__builtin_amdgcn_global_load_async_to_lds_b128)
#define HAVE_ASYNC_LDS 1
typedef __attribute__((address_space(1))) v4i* g_v4i_p;
typedef __attribute__((address_space(3))) v4i* l_v4i_p;
#endif

__device__ __forceinline__ void stage_b128(unsigned short* l, const unsigned short* g) {
#ifdef HAVE_ASYNC_LDS
  __builtin_amdgcn_global_load_async_to_lds_b128((g_v4i_p)g, (l_v4i_p)l, 0, 0);
#else
  *(f32x4*)l = *(const f32x4*)g;
#endif
}

__device__ __forceinline__ void wait_async() {
#ifdef HAVE_ASYNC_LDS
#if __has_builtin(__builtin_amdgcn_s_wait_asynccnt)
  __builtin_amdgcn_s_wait_asynccnt(0);
#else
  asm volatile("s_wait_asynccnt 0x0" ::: "memory");
#endif
#endif
}

// A-matrix fragment (16x32 bf16, M x K). row_base = &A[(m0 + lane%16)*ld + k0]
// Per ISA 7.12.2: lane holds row lane%16; elems 0..7 at K=half*8+e, 8..15 at K=16+half*8+e
__device__ __forceinline__ v16bf load_a_frag(const unsigned short* row_base, int half) {
  AF f;
  f.q[0] = *(const f32x4*)(row_base + half * 8);
  f.q[1] = *(const f32x4*)(row_base + 16 + half * 8);
  return f.v;
}

// B-matrix fragment (32x16 bf16, K x N). p = &B[k0*ldb + n0 + lane%16]
// Lane holds column N=lane%16; elems 0..7 at K=half*8+e, 8..15 at K=16+half*8+e
__device__ __forceinline__ v16bf load_b_frag(const unsigned short* p, int ld, int half) {
  union { v16bf v; unsigned short u[16]; } f;
#pragma unroll
  for (int e = 0; e < 8; ++e) {
    f.u[e]     = p[(half * 8 + e) * ld];
    f.u[e + 8] = p[(16 + half * 8 + e) * ld];
  }
  return f.v;
}

__device__ __forceinline__ v8f wmma_bf16(v16bf a, v16bf b, v8f c) {
  return __builtin_amdgcn_wmma_f32_16x16x32_bf16(false, a, false, b, (short)0, c, false, false);
}

// ---------------------------------------------------------------------------
// Problem constants
// ---------------------------------------------------------------------------
#define L_TOK   8192
#define D_MOD   1152
#define N_HEAD  16
#define DH      72
#define DH_QK   96    // K-dim pad for QK^T (multiple of 32)
#define DH_V    80    // N-dim pad for P*V  (multiple of 16)
#define N_IMG   8
#define SEQ     1024
#define QKV_N   (3 * D_MOD)

// ---------------------------------------------------------------------------
// float32 -> bf16 conversion
// ---------------------------------------------------------------------------
__global__ __launch_bounds__(256) void cvt_f32_bf16(const float* __restrict__ in,
                                                    unsigned short* __restrict__ out, int n) {
  int i = blockIdx.x * 256 + threadIdx.x;
  if (i < n) out[i] = f2bf(in[i]);
}

// ---------------------------------------------------------------------------
// GEMM: C(MxN,f32) = A(MxK,bf16) * B(KxN,bf16) + bias(N)
// Block: 256 thr = 8 waves, each wave a 16(M) x 64(N) strip (block 128x64).
// B tile (32x64) is async-staged into double-buffered LDS and shared by all
// 8 waves; A frags are register-prefetched one k-step ahead.
// ---------------------------------------------------------------------------
#define BK 32
#define BN 64
#define BS_STRIDE 72   // 64 + 8 pad shorts; 144B row stride (16B aligned)

__global__ __launch_bounds__(256) void gemm_bf16_bias(const unsigned short* __restrict__ A,
                                                      const unsigned short* __restrict__ B,
                                                      const float* __restrict__ bias,
                                                      float* __restrict__ C,
                                                      int M, int N, int K) {
  __shared__ __align__(16) unsigned short Bs[2][BK * BS_STRIDE];

  const int tid  = threadIdx.x;
  const int lane = tid & 31;
  const int wave = tid >> 5;
  const int half = lane >> 4;
  const int col  = lane & 15;
  const int m0 = blockIdx.y * 128 + wave * 16;
  const int n0 = blockIdx.x * BN;

  // staging indices: each thread moves one 16B chunk per k-step
  const int brow   = tid >> 3;        // 0..31
  const int bchunk = (tid & 7) * 8;   // short offset 0,8,..,56
  const unsigned short* bg = B + (size_t)brow * N + n0 + bchunk;
  unsigned short* bl0 = &Bs[0][brow * BS_STRIDE + bchunk];
  unsigned short* bl1 = bl0 + BK * BS_STRIDE;

  // prologue: stage k-tile 0 into buffer 0
  stage_b128(bl0, bg);

  const unsigned short* arow = A + (size_t)(m0 + col) * K;
  v16bf a_cur = load_a_frag(arow, half);

  const unsigned short* bs0 = &Bs[0][col];
  const unsigned short* bs1 = bs0 + BK * BS_STRIDE;

  v8f acc[4] = {};
  const int nsteps = K / BK;
  for (int kk = 0; kk < nsteps; ++kk) {
    wait_async();
    __syncthreads();                        // tile kk resident in Bs[kk&1]

    if (kk + 1 < nsteps)                    // stream next tile into other buffer
      stage_b128((kk & 1) ? bl0 : bl1, bg + (size_t)(kk + 1) * BK * N);

    const unsigned short* bs = (kk & 1) ? bs1 : bs0;
    v16bf b0 = load_b_frag(bs, BS_STRIDE, half);
    v16bf b1 = load_b_frag(bs + 16, BS_STRIDE, half);
    v16bf b2 = load_b_frag(bs + 32, BS_STRIDE, half);
    v16bf b3 = load_b_frag(bs + 48, BS_STRIDE, half);

    v16bf a = a_cur;
    if (kk + 1 < nsteps) a_cur = load_a_frag(arow + (kk + 1) * BK, half);

    acc[0] = wmma_bf16(a, b0, acc[0]);
    acc[1] = wmma_bf16(a, b1, acc[1]);
    acc[2] = wmma_bf16(a, b2, acc[2]);
    acc[3] = wmma_bf16(a, b3, acc[3]);
  }

  const int rowbase = m0 + 8 * half;
#pragma unroll
  for (int t = 0; t < 4; ++t) {
    int c = n0 + t * 16 + col;
    float bv = bias[c];
#pragma unroll
    for (int v = 0; v < 8; ++v)
      C[(size_t)(rowbase + v) * N + c] = acc[t][v] + bv;
  }
}

// ---------------------------------------------------------------------------
// RoPE + split/transpose/pad into attention layouts:
//   Qp[ih][s][0..95]  (bf16, cols 72..95 zero)
//   Kt[ih][d][s]      (bf16, rows 72..95 zero)  -- pre-transposed for B-frags
//   Vp[ih][s][0..79]  (bf16, cols 72..79 zero)
// ---------------------------------------------------------------------------
__global__ __launch_bounds__(256) void rope_split(const float* __restrict__ qkv,
                                                  const float* __restrict__ fcos,
                                                  const float* __restrict__ fsin,
                                                  unsigned short* __restrict__ Qp,
                                                  unsigned short* __restrict__ Kt,
                                                  unsigned short* __restrict__ Vp) {
  int tid   = blockIdx.x * 256 + threadIdx.x;     // < 8192*16
  int head  = tid & 15;
  int token = tid >> 4;
  int img   = token >> 10;
  int s     = token & (SEQ - 1);
  int ih    = img * N_HEAD + head;

  const float* qrow = qkv + (size_t)token * QKV_N + head * DH;
  const float* krow = qrow + D_MOD;
  const float* vrow = qrow + 2 * D_MOD;
  const float* cr = fcos + (size_t)token * (DH / 2);
  const float* sr = fsin + (size_t)token * (DH / 2);

  unsigned short* qd = Qp + ((size_t)ih * SEQ + s) * DH_QK;
  unsigned short* kd = Kt + ((size_t)ih * DH_QK) * SEQ + s;
  unsigned short* vd = Vp + ((size_t)ih * SEQ + s) * DH_V;

  for (int i = 0; i < DH / 2; ++i) {
    float c = cr[i], sn = sr[i];
    float qr = qrow[2 * i], qi = qrow[2 * i + 1];
    qd[2 * i]     = f2bf(qr * c - qi * sn);
    qd[2 * i + 1] = f2bf(qr * sn + qi * c);
    float kr = krow[2 * i], ki = krow[2 * i + 1];
    kd[(size_t)(2 * i) * SEQ]     = f2bf(kr * c - ki * sn);
    kd[(size_t)(2 * i + 1) * SEQ] = f2bf(kr * sn + ki * c);
  }
  for (int d = DH; d < DH_QK; ++d) { qd[d] = 0; kd[(size_t)d * SEQ] = 0; }
  for (int d = 0; d < DH; ++d) vd[d] = f2bf(vrow[d]);
  for (int d = DH; d < DH_V; ++d) vd[d] = 0;
}

// ---------------------------------------------------------------------------
// Flash attention per (img, head): block = 4 waves x 16 q-rows = 64 q rows.
// K/V tiles for each 32-key step are async-staged into double-buffered LDS
// (shared by the 4 waves); online softmax in registers; P relayout C->A via
// a 1KB per-wave LDS tile (within-wave LDS ops are in-order per the ISA).
// ---------------------------------------------------------------------------
#define KT_STRIDE 40   // 32 + 8 pad shorts (80B rows)
#define VS_STRIDE 88   // 80 + 8 pad shorts (176B rows)

__global__ __launch_bounds__(128) void attn_kernel(const unsigned short* __restrict__ Qp,
                                                   const unsigned short* __restrict__ Kt,
                                                   const unsigned short* __restrict__ Vp,
                                                   unsigned short* __restrict__ attnb,
                                                   float scale) {
  __shared__ __align__(16) unsigned short KtS[2][DH_QK * KT_STRIDE];  // 96x32 tile
  __shared__ __align__(16) unsigned short VS[2][32 * VS_STRIDE];      // 32x80 tile
  __shared__ __align__(16) unsigned short pbuf[4][16 * 32];

  const int tid  = threadIdx.x;
  const int lane = tid & 31;
  const int wave = tid >> 5;
  const int half = lane >> 4;
  const int col  = lane & 15;
  const int ih = blockIdx.x >> 4;                   // 0..127 = img*16+head
  const int q0 = (blockIdx.x & 15) * 64 + wave * 16;

  const unsigned short* Qh  = Qp + (size_t)ih * SEQ * DH_QK;
  const unsigned short* Kth = Kt + (size_t)ih * DH_QK * SEQ;
  const unsigned short* Vh  = Vp + (size_t)ih * SEQ * DH_V;

  // Preload the wave's Q tile: 3 A-frags covering K = 0..95
  const unsigned short* qrow = Qh + (size_t)(q0 + col) * DH_QK;
  v16bf qf[3];
#pragma unroll
  for (int kk = 0; kk < 3; ++kk) qf[kk] = load_a_frag(qrow + kk * 32, half);

  v8f o[5] = {};
  float mrun[8], lrun[8];
#pragma unroll
  for (int v = 0; v < 8; ++v) { mrun[v] = -1e30f; lrun[v] = 0.f; }

  unsigned short* pb = pbuf[wave];

  // --- staging: K tile = 96 rows x 32 cols (384 chunks),
  //              V tile = 32 rows x 80 cols (320 chunks), 16B each ---
  auto stage_tile = [&](int j0, int buf) {
    for (int c = tid; c < 384; c += 128) {
      int r = c >> 2, ch = (c & 3) * 8;
      stage_b128(&KtS[buf][r * KT_STRIDE + ch], Kth + (size_t)r * SEQ + j0 + ch);
    }
    for (int c = tid; c < 320; c += 128) {
      int r = c / 10, ch = (c - r * 10) * 8;
      stage_b128(&VS[buf][r * VS_STRIDE + ch], Vh + (size_t)(j0 + r) * DH_V + ch);
    }
  };

  stage_tile(0, 0);   // prologue

  const int njs = SEQ / 32;
  for (int jj = 0; jj < njs; ++jj) {
    wait_async();
    __syncthreads();                      // tile jj resident
    if (jj + 1 < njs) stage_tile((jj + 1) * 32, (jj + 1) & 1);

    const unsigned short* kts = KtS[jj & 1];
    const unsigned short* vs  = VS[jj & 1];

    // scores for 32 keys: two 16x16 tiles
    v8f s0 = {}, s1 = {};
#pragma unroll
    for (int kk = 0; kk < 3; ++kk) {
      const unsigned short* kp = kts + (kk * 32) * KT_STRIDE + col;
      s0 = wmma_bf16(qf[kk], load_b_frag(kp, KT_STRIDE, half), s0);
      s1 = wmma_bf16(qf[kk], load_b_frag(kp + 16, KT_STRIDE, half), s1);
    }

    // row-wise online softmax; C-layout: VGPR v = row v+8*half, lane = column
    float pm[8];
#pragma unroll
    for (int v = 0; v < 8; ++v) {
      s0[v] *= scale; s1[v] *= scale;
      float m = fmaxf(s0[v], s1[v]);
      m = fmaxf(m, __shfl_xor(m, 1));
      m = fmaxf(m, __shfl_xor(m, 2));
      m = fmaxf(m, __shfl_xor(m, 4));
      m = fmaxf(m, __shfl_xor(m, 8));
      pm[v] = m;
    }
    float alpha[8];
#pragma unroll
    for (int v = 0; v < 8; ++v) {
      float nm = fmaxf(mrun[v], pm[v]);
      alpha[v] = __expf(mrun[v] - nm);
      mrun[v] = nm;
      float p0 = __expf(s0[v] - nm);
      float p1 = __expf(s1[v] - nm);
      s0[v] = p0; s1[v] = p1;
      float rs = p0 + p1;
      rs += __shfl_xor(rs, 1);
      rs += __shfl_xor(rs, 2);
      rs += __shfl_xor(rs, 4);
      rs += __shfl_xor(rs, 8);
      lrun[v] = lrun[v] * alpha[v] + rs;
    }
#pragma unroll
    for (int t = 0; t < 5; ++t)
#pragma unroll
      for (int v = 0; v < 8; ++v) o[t][v] *= alpha[v];

    // P: C-layout -> per-wave LDS tile -> A-frag layout
#pragma unroll
    for (int v = 0; v < 8; ++v) {
      int r = v + 8 * half;
      pb[r * 32 + col]      = f2bf(s0[v]);
      pb[r * 32 + 16 + col] = f2bf(s1[v]);
    }
    v16bf pf = load_a_frag(pb + col * 32, half);

    // o += P (16x32) * V (32x80)
#pragma unroll
    for (int t = 0; t < 5; ++t)
      o[t] = wmma_bf16(pf, load_b_frag(vs + col + t * 16, VS_STRIDE, half), o[t]);
  }

  // finalize: o /= l, scatter valid head dims (0..71) as bf16
  const int img = ih >> 4, head = ih & 15;
  const size_t rowbase = (size_t)img * SEQ + q0 + 8 * half;
#pragma unroll
  for (int v = 0; v < 8; ++v) {
    float inv = 1.f / lrun[v];
    size_t outrow = (rowbase + v) * D_MOD + (size_t)head * DH;
#pragma unroll
    for (int t = 0; t < 5; ++t) {
      int c = t * 16 + col;
      if (c < DH) attnb[outrow + c] = f2bf(o[t][v] * inv);
    }
  }
}

// ---------------------------------------------------------------------------
// Host-side orchestration
// ---------------------------------------------------------------------------
extern "C" void kernel_launch(void* const* d_in, const int* in_sizes, int n_in,
                              void* d_out, int out_size, void* d_ws, size_t ws_size,
                              hipStream_t stream) {
  const float* x    = (const float*)d_in[0];
  const float* wqkv = (const float*)d_in[1];
  const float* bqkv = (const float*)d_in[2];
  const float* wo   = (const float*)d_in[3];
  const float* bo   = (const float*)d_in[4];
  const float* fcos = (const float*)d_in[5];
  const float* fsin = (const float*)d_in[6];
  float* out = (float*)d_out;

  // Workspace layout (all 16B aligned)
  char* ws = (char*)d_ws;
  const size_t sz_qkv   = (size_t)L_TOK * QKV_N * sizeof(float);             // 113246208
  const size_t sz_xb    = (size_t)L_TOK * D_MOD * 2;                         // 18874368
  const size_t sz_wqkvb = (size_t)D_MOD * QKV_N * 2;                         // 7962624
  const size_t sz_wob   = (size_t)D_MOD * D_MOD * 2;                         // 2654208
  const size_t sz_q     = (size_t)N_IMG * N_HEAD * SEQ * DH_QK * 2;          // 25165824
  const size_t sz_kt    = sz_q;
  float*          qkv   = (float*)ws;
  unsigned short* xb    = (unsigned short*)(ws + sz_qkv);
  unsigned short* wqkvb = (unsigned short*)(ws + sz_qkv + sz_xb);
  unsigned short* wob   = (unsigned short*)(ws + sz_qkv + sz_xb + sz_wqkvb);
  unsigned short* Qpb   = (unsigned short*)(ws + sz_qkv + sz_xb + sz_wqkvb + sz_wob);
  unsigned short* Ktb   = (unsigned short*)(ws + sz_qkv + sz_xb + sz_wqkvb + sz_wob + sz_q);
  unsigned short* Vpb   = (unsigned short*)(ws + sz_qkv + sz_xb + sz_wqkvb + sz_wob + sz_q + sz_kt);
  unsigned short* attnb = (unsigned short*)ws;   // aliases qkv (dead after rope_split)

  // 1) bf16 conversions
  {
    int n = L_TOK * D_MOD;
    cvt_f32_bf16<<<(n + 255) / 256, 256, 0, stream>>>(x, xb, n);
    n = D_MOD * QKV_N;
    cvt_f32_bf16<<<(n + 255) / 256, 256, 0, stream>>>(wqkv, wqkvb, n);
    n = D_MOD * D_MOD;
    cvt_f32_bf16<<<(n + 255) / 256, 256, 0, stream>>>(wo, wob, n);
  }

  // 2) qkv = x @ wqkv + bqkv   (8192 x 3456, K=1152)
  gemm_bf16_bias<<<dim3(QKV_N / BN, L_TOK / 128), 256, 0, stream>>>(
      xb, wqkvb, bqkv, qkv, L_TOK, QKV_N, D_MOD);

  // 3) RoPE + split into attention layouts
  rope_split<<<(L_TOK * N_HEAD) / 256, 256, 0, stream>>>(qkv, fcos, fsin, Qpb, Ktb, Vpb);

  // 4) flash attention (8 img x 16 heads x 16 q-blocks of 64 rows)
  attn_kernel<<<N_IMG * N_HEAD * (SEQ / 64), 128, 0, stream>>>(
      Qpb, Ktb, Vpb, attnb, 0.11785113019775793f /* 72^-0.5 */);

  // 5) out = attn @ wo + bo    (8192 x 1152, K=1152)
  gemm_bf16_bias<<<dim3(D_MOD / BN, L_TOK / 128), 256, 0, stream>>>(
      attnb, wob, bo, out, L_TOK, D_MOD, D_MOD);
}
